// Classifier_41927470744090
// MI455X (gfx1250) — matrix operand order repaired and checked by reading, not compiled, not run on hardware
//
#include <hip/hip_runtime.h>
#include <hip/hip_bf16.h>
#include <stdint.h>

// ---------------------------------------------------------------------------
// R-GCN (2 layers) + mean pool + linear + softmax for MI455X (gfx1250).
// Matmuls in bf16 via v_wmma_f32_16x16x32_bf16, f32 accumulate.
// Edge messages computed as 64-edge tiles sorted by relation (4 M-tiles per
// block share every B fragment); A tiles staged in LDS via async
// global->LDS copies (GLOBAL_LOAD_ASYNC_TO_LDS_B128, ASYNCcnt);
// scatter via global f32 atomics with 32-bit scaled offsets.
// Weights pre-transposed to [R][O][K] bf16 so each B fragment is two
// 16B loads from one K row per lane (CDNA5 ISA dense-B layout).
// ---------------------------------------------------------------------------

typedef __attribute__((ext_vector_type(16))) __bf16 v16bf;
typedef __attribute__((ext_vector_type(8)))  __bf16 v8bf;
typedef __attribute__((ext_vector_type(8)))  float  v8f;

#define N_NODES   100000
#define N_EDGES   600000
#define N_RELS    8
#define IN_DIM    128
#define HID       256
#define N_CLASSES 32
#define N_GRAPHS  50

#define MTILE       64                            // edges per block
#define EPAD        (N_EDGES + N_RELS * MTILE)    // padded sorted-edge capacity
#define EDGE_BLOCKS (N_EDGES / MTILE + N_RELS)    // upper bound on tiles
#define TRASH_ROW   N_NODES                       // sentinel edges dump here

#if __has_builtin(__builtin_amdgcn_global_load_async_to_lds_b128) && \
    __has_builtin(__builtin_amdgcn_s_wait_asynccnt)
#define USE_ASYNC_GATHER 1
// Builtin prototype (from hipcc diagnostic): param0 = vector_size(16) int in
// addrspace(1) ("__device__"), param1 = same vector in addrspace(3), +2 imms.
typedef int v4i_b128 __attribute__((vector_size(16)));
typedef __attribute__((address_space(1))) v4i_b128* gptr_b128;
typedef __attribute__((address_space(3))) v4i_b128* lptr_b128;
#else
#define USE_ASYNC_GATHER 0
#endif

__device__ __forceinline__ unsigned short f2bf(float f) {
  unsigned int u = __float_as_uint(f);
  u += 0x7fffu + ((u >> 16) & 1u);                // round-to-nearest-even
  return (unsigned short)(u >> 16);
}

// ---------------------------- small utility kernels ------------------------

__global__ void k_zero_u32(unsigned int* p, long n) {
  long i = (long)blockIdx.x * blockDim.x + threadIdx.x;
  if (i < n) p[i] = 0u;
}

__global__ void k_zero_u128(uint4* p, long n4) {           // n4 = count of uint4
  long i = (long)blockIdx.x * blockDim.x + threadIdx.x;
  if (i < n4) p[i] = make_uint4(0u, 0u, 0u, 0u);
}

__global__ void k_cvt_bf16(const float* __restrict__ in,
                           unsigned short* __restrict__ out, long n) {
  long i = (long)blockIdx.x * blockDim.x + threadIdx.x;
  if (i < n) out[i] = f2bf(in[i]);
}

// W[B][K][O] (f32) -> Wt[B][O][K] (bf16 bits)
__global__ void k_transpose(const float* __restrict__ W,
                            unsigned short* __restrict__ Wt,
                            int B, int Kd, int Od) {
  long i = (long)blockIdx.x * blockDim.x + threadIdx.x;
  long total = (long)B * Kd * Od;
  if (i >= total) return;
  int o = (int)(i % Od);
  long t = i / Od;
  int k = (int)(t % Kd);
  int b = (int)(t / Kd);
  Wt[((long)b * Od + o) * Kd + k] = f2bf(W[i]);
}

__global__ void k_count_rels(const int* __restrict__ rel, unsigned int* cnt) {
  int e = blockIdx.x * blockDim.x + threadIdx.x;
  if (e < N_EDGES) atomicAdd(&cnt[rel[e]], 1u);
}

__global__ void k_plan(const unsigned int* __restrict__ cnt, int* pad_off) {
  if (blockIdx.x == 0 && threadIdx.x == 0) {
    int b = 0;
    pad_off[0] = 0;
    for (int r = 0; r < N_RELS; ++r) {
      b += (((int)cnt[r] + MTILE - 1) / MTILE) * MTILE;
      pad_off[r + 1] = b;
    }
  }
}

__global__ void k_fill_sentinel(int* src_s, int* dst_s) {
  int i = blockIdx.x * blockDim.x + threadIdx.x;
  if (i < EPAD) { src_s[i] = 0; dst_s[i] = TRASH_ROW; }
}

__global__ void k_scatter_edges(const int* __restrict__ src,
                                const int* __restrict__ dst,
                                const int* __restrict__ rel,
                                const int* __restrict__ pad_off,
                                unsigned int* cursor,
                                int* src_s, int* dst_s) {
  int e = blockIdx.x * blockDim.x + threadIdx.x;
  if (e >= N_EDGES) return;
  int r = rel[e];
  int pos = pad_off[r] + (int)atomicAdd(&cursor[r], 1u);
  src_s[pos] = src[e];
  dst_s[pos] = dst[e];
}

// ---------------------------- WMMA GEMM kernels ----------------------------

// Build a 16x32 bf16 A fragment per ISA layout: row m = lane&15,
// elems 0..7 -> K = kbase..kbase+7, elems 8..15 -> K = kbase+16..kbase+23,
// kbase = 8*(lane>=16).
__device__ __forceinline__ v16bf load_a_frag(const unsigned short* p) {
  const v8bf lo = *(const v8bf*)p;
  const v8bf hi = *(const v8bf*)(p + 16);
  v16bf f;
#pragma unroll
  for (int j = 0; j < 8; ++j) { f[j] = lo[j]; f[8 + j] = hi[j]; }
  return f;
}

// One block = 64 edges (4 M-tiles) x 256 outputs (16 waves x 16 cols each).
template <int K>
__global__ void __launch_bounds__(512) k_edge_gemm(
    const unsigned short* __restrict__ hbf,   // [N][K] bf16
    const unsigned short* __restrict__ Wt,    // [R][HID][K] bf16 (transposed)
    const int* __restrict__ src_s,
    const int* __restrict__ dst_s,
    const int* __restrict__ pad_off,          // [R+1]
    float* __restrict__ acc)                  // [(N+MTILE)][HID]
{
  __shared__ __align__(16) unsigned short at[MTILE * K];
  const int e0 = blockIdx.x * MTILE;
  if (e0 >= pad_off[N_RELS]) return;          // uniform across block
  int r = 0;
#pragma unroll
  for (int i = 1; i < N_RELS; ++i) r += (e0 >= pad_off[i]) ? 1 : 0;

  const int tid  = threadIdx.x;
  const int wave = tid >> 5;
  const int lane = tid & 31;
  const int n0   = wave * 16;
  const int n    = lane & 15;                 // B/D column, A row
  const int half = lane >> 4;
  const int koff = half * 8;                  // A K-offset per lane half

  // Kick the weight stream into the caches while we stage A in LDS.
  const unsigned short* wrow = Wt + ((unsigned)r * HID + (unsigned)(n0 + n)) * K;
  __builtin_prefetch(wrow, 0, 2);
  __builtin_prefetch(wrow + K / 2, 0, 2);

  // Cooperative gather of A tile: MTILE rows x K bf16, 16B per chunk.
  // Uniform trip count across the block (no divergence around async ops).
  constexpr int CH = MTILE * K / 8;
#pragma unroll
  for (int c0 = 0; c0 < CH; c0 += 512) {
    const int c   = c0 + tid;
    const int row = c / (K / 8);
    const int kk  = (c % (K / 8)) * 8;
    const unsigned node = (unsigned)src_s[e0 + row];
#if USE_ASYNC_GATHER
    __builtin_amdgcn_global_load_async_to_lds_b128(
        (gptr_b128)(void*)(hbf + node * K + kk),
        (lptr_b128)(void*)&at[row * K + kk], 0, 0);
#else
    *(uint4*)&at[row * K + kk] = *(const uint4*)&hbf[node * K + kk];
#endif
  }
#if USE_ASYNC_GATHER
  __builtin_amdgcn_s_wait_asynccnt(0);        // our LDS writes landed
#endif
  __syncthreads();                            // everyone's LDS writes visible

  v8f cacc[4] = {{}, {}, {}, {}};

#pragma unroll
  for (int ks = 0; ks < K / 32; ++ks) {
    const int k0 = ks * 32;
    const v16bf bfrag = load_a_frag(wrow + k0 + koff);  // same K order as A
#pragma unroll
    for (int t = 0; t < 4; ++t) {
      const v16bf afrag = load_a_frag(&at[(t * 16 + n) * K + k0 + koff]);
      cacc[t] = __builtin_amdgcn_wmma_f32_16x16x32_bf16(
          false, afrag, false, bfrag, (short)0, cacc[t], false, false);
    }
  }

  // D layout: vgpr i, lane -> m = i + 8*half, n = lane&15. Scatter-add.
#pragma unroll
  for (int t = 0; t < 4; ++t) {
#pragma unroll
    for (int i = 0; i < 8; ++i) {
      const int m = i + half * 8;
      const unsigned d = (unsigned)dst_s[e0 + t * 16 + m];
      atomicAdd(acc + (d * HID + (unsigned)(n0 + n)), cacc[t][i]);
    }
  }
}

// acc[m0..m0+31][:] += x[m0..m0+31][:] @ L  (L pre-transposed [HID][K])
template <int K>
__global__ void __launch_bounds__(512) k_selfloop_gemm(
    const unsigned short* __restrict__ xbf,   // [N][K]
    const unsigned short* __restrict__ Lt,    // [HID][K]
    float* __restrict__ acc)
{
  const unsigned m0 = blockIdx.x * 32;        // N_NODES % 32 == 0
  const int tid = threadIdx.x;
  const int wave = tid >> 5, lane = tid & 31;
  const int n0 = wave * 16, n = lane & 15, half = lane >> 4, koff = half * 8;

  v8f c0 = {}, c1 = {};
  const unsigned short* wrow   = Lt + (unsigned)(n0 + n) * K;
  const unsigned short* a0base = xbf + (m0 + (unsigned)n) * K;
  const unsigned short* a1base = xbf + (m0 + 16u + (unsigned)n) * K;

#pragma unroll
  for (int ks = 0; ks < K / 32; ++ks) {
    const int k0 = ks * 32;
    const v16bf bfrag = load_a_frag(wrow + k0 + koff);
    const v16bf af0 = load_a_frag(a0base + k0 + koff);
    const v16bf af1 = load_a_frag(a1base + k0 + koff);
    c0 = __builtin_amdgcn_wmma_f32_16x16x32_bf16(false, af0, false, bfrag,
                                                 (short)0, c0, false, false);
    c1 = __builtin_amdgcn_wmma_f32_16x16x32_bf16(false, af1, false, bfrag,
                                                 (short)0, c1, false, false);
  }
#pragma unroll
  for (int i = 0; i < 8; ++i) {
    const unsigned m = (unsigned)(i + half * 8);
    acc[(m0 + m) * HID + (unsigned)(n0 + n)] += c0[i];
    acc[(m0 + 16u + m) * HID + (unsigned)(n0 + n)] += c1[i];
  }
}

// ---------------------------- epilogue kernels -----------------------------

__global__ void k_finalize_relu_bf16(const float* __restrict__ acc,
                                     const float* __restrict__ bias,
                                     unsigned short* __restrict__ out) {
  long i = (long)blockIdx.x * blockDim.x + threadIdx.x;
  if (i >= (long)N_NODES * HID) return;
  int o = (int)(i % HID);
  float v = acc[i] + bias[o];
  out[i] = f2bf(v > 0.f ? v : 0.f);
}

__global__ void k_finalize_pool(const float* __restrict__ acc,
                                const float* __restrict__ bias,
                                const int* __restrict__ gid,
                                float* __restrict__ hg, int* __restrict__ gcnt) {
  long i = (long)blockIdx.x * blockDim.x + threadIdx.x;
  if (i >= (long)N_NODES * HID) return;
  int o = (int)(i % HID);
  int nrow = (int)(i / HID);
  float v = acc[i] + bias[o];
  v = v > 0.f ? v : 0.f;
  int g = gid[nrow];
  atomicAdd(&hg[g * HID + o], v);
  if (o == 0) atomicAdd(&gcnt[g], 1);
}

// 50 blocks x 32 threads (one wave32); softmax via lane shuffles.
__global__ void k_classifier(const float* __restrict__ hg,
                             const int* __restrict__ gcnt,
                             const float* __restrict__ Wc,
                             const float* __restrict__ bc,
                             float* __restrict__ out) {
  const int g = blockIdx.x;
  const int c = threadIdx.x;                  // 0..31
  const int cn = gcnt[g];
  const float inv = 1.0f / (float)(cn > 0 ? cn : 1);
  float s = 0.f;
  for (int k = 0; k < HID; ++k) s += hg[g * HID + k] * Wc[k * N_CLASSES + c];
  s = s * inv + bc[c];
  float m = s;
#pragma unroll
  for (int off = 16; off >= 1; off >>= 1) m = fmaxf(m, __shfl_xor(m, off, 32));
  float e = __expf(s - m);
  float sum = e;
#pragma unroll
  for (int off = 16; off >= 1; off >>= 1) sum += __shfl_xor(sum, off, 32);
  out[g * N_CLASSES + c] = e / sum;
}

// ---------------------------------------------------------------------------

extern "C" void kernel_launch(void* const* d_in, const int* in_sizes, int n_in,
                              void* d_out, int out_size, void* d_ws, size_t ws_size,
                              hipStream_t stream) {
  (void)in_sizes; (void)n_in; (void)out_size; (void)ws_size;

  const float* h   = (const float*)d_in[0];
  const int*   src = (const int*)d_in[1];
  const int*   dst = (const int*)d_in[2];
  const int*   rel = (const int*)d_in[3];
  const int*   gid = (const int*)d_in[4];
  const float* W1  = (const float*)d_in[5];
  const float* l1  = (const float*)d_in[6];
  const float* b1  = (const float*)d_in[7];
  const float* W2  = (const float*)d_in[8];
  const float* l2  = (const float*)d_in[9];
  const float* b2  = (const float*)d_in[10];
  const float* Wc  = (const float*)d_in[11];
  const float* bc  = (const float*)d_in[12];
  float* out = (float*)d_out;

  char* p = (char*)d_ws;
  auto alloc = [&](size_t bytes) -> char* {
    char* q = p;
    p += (bytes + 255) & ~(size_t)255;
    return q;
  };
  unsigned short* hbf  = (unsigned short*)alloc((size_t)N_NODES * IN_DIM * 2);
  unsigned short* h1bf = (unsigned short*)alloc((size_t)N_NODES * HID * 2);
  unsigned short* W1t  = (unsigned short*)alloc((size_t)N_RELS * HID * IN_DIM * 2);
  unsigned short* l1t  = (unsigned short*)alloc((size_t)HID * IN_DIM * 2);
  unsigned short* W2t  = (unsigned short*)alloc((size_t)N_RELS * HID * HID * 2);
  unsigned short* l2t  = (unsigned short*)alloc((size_t)HID * HID * 2);
  float*        acc    = (float*)alloc((size_t)(N_NODES + MTILE) * HID * 4);
  int*          src_s  = (int*)alloc((size_t)EPAD * 4);
  int*          dst_s  = (int*)alloc((size_t)EPAD * 4);
  unsigned int* cnt    = (unsigned int*)alloc(64);
  unsigned int* cur    = (unsigned int*)alloc(64);
  int*          padoff = (int*)alloc(64);
  float*        hg     = (float*)alloc((size_t)N_GRAPHS * HID * 4);
  int*          gcnt   = (int*)alloc(256);

  const int T = 256;
  auto blocks = [](long n, int t) { return (int)((n + t - 1) / t); };
  const long accN = (long)(N_NODES + MTILE) * HID;

  // --- precision conversion + weight transposes (bf16) ---
  k_cvt_bf16<<<blocks((long)N_NODES * IN_DIM, T), T, 0, stream>>>(h, hbf, (long)N_NODES * IN_DIM);
  k_transpose<<<blocks((long)N_RELS * IN_DIM * HID, T), T, 0, stream>>>(W1, W1t, N_RELS, IN_DIM, HID);
  k_transpose<<<blocks((long)IN_DIM * HID, T), T, 0, stream>>>(l1, l1t, 1, IN_DIM, HID);
  k_transpose<<<blocks((long)N_RELS * HID * HID, T), T, 0, stream>>>(W2, W2t, N_RELS, HID, HID);
  k_transpose<<<blocks((long)HID * HID, T), T, 0, stream>>>(l2, l2t, 1, HID, HID);

  // --- counting sort of edges by relation, padded to MTILE-edge tiles ---
  k_zero_u32<<<1, 64, 0, stream>>>(cnt, 16);
  k_zero_u32<<<1, 64, 0, stream>>>(cur, 16);
  k_count_rels<<<blocks(N_EDGES, T), T, 0, stream>>>(rel, cnt);
  k_plan<<<1, 1, 0, stream>>>(cnt, padoff);
  k_fill_sentinel<<<blocks(EPAD, T), T, 0, stream>>>(src_s, dst_s);
  k_scatter_edges<<<blocks(N_EDGES, T), T, 0, stream>>>(src, dst, rel, padoff, cur, src_s, dst_s);

  // --- layer 1 ---
  k_zero_u128<<<blocks(accN / 4, T), T, 0, stream>>>((uint4*)acc, accN / 4);
  k_edge_gemm<IN_DIM><<<EDGE_BLOCKS, 512, 0, stream>>>(hbf, W1t, src_s, dst_s, padoff, acc);
  k_selfloop_gemm<IN_DIM><<<N_NODES / 32, 512, 0, stream>>>(hbf, l1t, acc);
  k_finalize_relu_bf16<<<blocks((long)N_NODES * HID, T), T, 0, stream>>>(acc, b1, h1bf);

  // --- layer 2 (reuse acc) ---
  k_zero_u128<<<blocks(accN / 4, T), T, 0, stream>>>((uint4*)acc, accN / 4);
  k_edge_gemm<HID><<<EDGE_BLOCKS, 512, 0, stream>>>(h1bf, W2t, src_s, dst_s, padoff, acc);
  k_selfloop_gemm<HID><<<N_NODES / 32, 512, 0, stream>>>(h1bf, l2t, acc);

  // --- mean pooling + classifier + softmax ---
  k_zero_u32<<<blocks((long)N_GRAPHS * HID, T), T, 0, stream>>>((unsigned int*)hg, (long)N_GRAPHS * HID);
  k_zero_u32<<<1, 64, 0, stream>>>((unsigned int*)gcnt, 64);
  k_finalize_pool<<<blocks((long)N_NODES * HID, T), T, 0, stream>>>(acc, b2, gid, hg, gcnt);
  k_classifier<<<N_GRAPHS, 32, 0, stream>>>(hg, gcnt, Wc, bc, out);
}
